// MultiHeadAttention_21543555957442
// MI455X (gfx1250) — matrix operand order repaired and checked by reading, not compiled, not run on hardware
//
#include <hip/hip_runtime.h>
#include <hip/hip_bf16.h>
#include <math.h>

typedef __attribute__((ext_vector_type(16))) __bf16 v16bf;
typedef __attribute__((ext_vector_type(8)))  float  v8f;
typedef __attribute__((ext_vector_type(4)))  int    v4i;

namespace {
constexpr int   kS     = 2048;  // sequence length
constexpr int   kH     = 16;    // heads
constexpr int   kDK    = 64;    // key/query head dim
constexpr int   kDV    = 64;    // value head dim
constexpr int   kKT    = 32;    // keys processed per iteration
constexpr int   kKP    = 72;    // K-stage row pitch in f32 (64 + 8 pad)
constexpr int   kVP    = 40;    // V^T-stage row pitch in bf16 (32 + 8 pad)
constexpr int   kSteps = kS / kKT;
constexpr float kScale = 0.125f; // 1/sqrt(64)
}

// CDNA5 async global->LDS copy path (tracked on ASYNCcnt), with fallback.
#if defined(__gfx1250__) && __has_builtin(__builtin_amdgcn_global_load_async_to_lds_b128)
#define FA_ASYNC_LDS 1
#else
#define FA_ASYNC_LDS 0
#endif

#if __has_builtin(__builtin_amdgcn_s_wait_asynccnt)
#define FA_WAIT_ASYNC() __builtin_amdgcn_s_wait_asynccnt(0)
#else
#define FA_WAIT_ASYNC() asm volatile("s_wait_asynccnt 0" ::: "memory")
#endif

// Stage one 32-key block: K tile verbatim (fp32; async DMA to LDS when
// available), V tile transposed + converted to bf16 (manual; async copies
// cannot transform data). 128 threads x 4 float4 chunks each.
__device__ __forceinline__
void stage_tiles(int tid, const float* __restrict__ Kh,
                 const float* __restrict__ Vh, int k0,
                 float* kbuf, __bf16* vbuf)
{
#pragma unroll
    for (int r = 0; r < 4; ++r) {
        const int c   = r * 128 + tid;
        const int row = c >> 4;            // key row 0..31
        const int c4  = (c & 15) * 4;      // column within row
#if FA_ASYNC_LDS
        __builtin_amdgcn_global_load_async_to_lds_b128(
            (__attribute__((address_space(1))) v4i*)(Kh + (size_t)(k0 + row) * kDK + c4),
            (__attribute__((address_space(3))) v4i*)(kbuf + row * kKP + c4),
            /*offset=*/0, /*cpol=*/0);
#else
        const float4 kv = *(const float4*)(Kh + (size_t)(k0 + row) * kDK + c4);
        *(float4*)(kbuf + row * kKP + c4) = kv;
#endif
        const float4 vv = *(const float4*)(Vh + (size_t)(k0 + row) * kDV + c4);
        vbuf[(c4 + 0) * kVP + row] = (__bf16)vv.x;
        vbuf[(c4 + 1) * kVP + row] = (__bf16)vv.y;
        vbuf[(c4 + 2) * kVP + row] = (__bf16)vv.z;
        vbuf[(c4 + 3) * kVP + row] = (__bf16)vv.w;
    }
}

// Flash-attention forward. One wave owns a 16-query strip; 4 waves (64 queries)
// per workgroup share double-buffered LDS tiles of 32 keys.
//
// Score matmul computed transposed: S^T[key][query] = K_tile x Q^T so that the
// WMMA C-layout (query across lanes, key across VGPRs) coincides with the
// 16-bit A-fragment layout needed for the P x V matmul -- no transpose needed.
__global__ __launch_bounds__(128, 1)
void fa_fwd_kernel(const float* __restrict__ Kg,
                   const float* __restrict__ Qg,
                   const float* __restrict__ Vg,
                   float* __restrict__ Og)
{
    __shared__ float  kst[2][kKT * kKP];   // K tiles, fp32, [key][dk] padded
    __shared__ __bf16 vst[2][kDV * kVP];   // V tiles transposed, bf16, [dv][key]

    const int tid  = threadIdx.x;
    const int lane = tid & 31;
    const int wv   = tid >> 5;        // wave 0..3
    const int half = lane >> 4;       // 0: lanes 0-15, 1: lanes 16-31
    const int l16  = lane & 15;

    const int b = blockIdx.z;
    const int h = blockIdx.y;
    const size_t head = (size_t)(b * kH + h) * kS;
    const float* Kh = Kg + head * kDK;
    const float* Qh = Qg + head * kDK;
    const float* Vh = Vg + head * kDV;
    float*       Oh = Og + head * kDV;   // raw-reshape output == [B,H,S,DV] flat

    const int q0 = blockIdx.x * 64 + wv * 16;   // this wave's query base

    // Prologue: stage first key block into buffer 0.
    stage_tiles(tid, Kh, Vh, 0, kst[0], vst[0]);

    // ---- Q^T as WMMA B-fragments (held for the whole pass), pre-scaled ----
    // B-fragment (K=32, 16-bit): lanes 0-15 hold col N=l16, contraction K=0..15;
    // lanes 16-31 hold K=16..31 -> each lane reads 16 contiguous f32 of its row.
    v16bf qb0, qb1;
    {
        const float* qr = Qh + (size_t)(q0 + l16) * kDK + half * 16;
#pragma unroll
        for (int i = 0; i < 16; ++i) {
            qb0[i] = (__bf16)(qr[i]      * kScale);   // dk 0..31 chunk
            qb1[i] = (__bf16)(qr[32 + i] * kScale);   // dk 32..63 chunk
        }
    }

    // Output accumulators: 4 column tiles of 16 dv each (C-layout).
    v8f acc0 = {}, acc1 = {}, acc2 = {}, acc3 = {};
    float mrun = -INFINITY;   // running row max (per query == per lane)
    float lrun = 0.0f;        // running row sum

    for (int step = 0; step < kSteps; ++step) {
        const int cur = step & 1;
#if FA_ASYNC_LDS
        FA_WAIT_ASYNC();      // our async copies for buffer `cur` have landed
#endif
        // One barrier per step: proves (a) buffer `cur` fully staged by all
        // threads, and (b) all waves finished reading buffer `cur^1` last
        // iteration, so it is safe to overwrite below.
        __syncthreads();

        if (step + 1 < kSteps)
            stage_tiles(tid, Kh, Vh, (step + 1) * kKT, kst[cur ^ 1], vst[cur ^ 1]);

        const float*  kb = kst[cur];
        const __bf16* vb_base = vst[cur];

        // ---- scores: S^T = K_tile x Q^T, two 16-key sub-tiles, K-dim = 64 ----
        v8f sct[2];
#pragma unroll
        for (int t = 0; t < 2; ++t) {
            // A-fragment (16-bit, K=32): lanes 0-15 row M=l16 hold K 0..15,
            // lanes 16-31 hold K 16..31 -> contiguous run per chunk (matches Q^T B).
            const float* krow = kb + (t * 16 + l16) * kKP;
            v16bf ka0, ka1;
#pragma unroll
            for (int i = 0; i < 8; ++i) {
                ka0[i]     = (__bf16)krow[half * 16 + i];
                ka0[i + 8] = (__bf16)krow[half * 16 + 8 + i];
                ka1[i]     = (__bf16)krow[32 + half * 16 + i];
                ka1[i + 8] = (__bf16)krow[32 + half * 16 + 8 + i];
            }
            v8f s = {};
            s = __builtin_amdgcn_wmma_f32_16x16x32_bf16(false, ka0, false, qb0,
                                                        (short)0, s, false, false);
            s = __builtin_amdgcn_wmma_f32_16x16x32_bf16(false, ka1, false, qb1,
                                                        (short)0, s, false, false);
            sct[t] = s;   // lane = query, VGPR j = key (t*16 + j + half*8)
        }

        // ---- online softmax (per-query stats live per-lane) ----
        float tmax = -INFINITY;
#pragma unroll
        for (int j = 0; j < 8; ++j)
            tmax = fmaxf(tmax, fmaxf(sct[0][j], sct[1][j]));
        tmax = fmaxf(tmax, __shfl_xor(tmax, 16, 32));   // merge the two key-halves
        const float mnew = fmaxf(mrun, tmax);
        const float corr = __expf(mrun - mnew);
        mrun = mnew;

        float p0[8], p1[8];
        float rsum = 0.0f;
#pragma unroll
        for (int j = 0; j < 8; ++j) {
            p0[j] = __expf(sct[0][j] - mnew);
            p1[j] = __expf(sct[1][j] - mnew);
            rsum += p0[j] + p1[j];
        }
        rsum += __shfl_xor(rsum, 16, 32);
        lrun = lrun * corr + rsum;

        // Rescale accumulators: acc VGPR j is query row (j + half*8); broadcast
        // that query's correction factor from its owning lane.
#pragma unroll
        for (int j = 0; j < 8; ++j) {
            const float rc = __shfl(corr, j + (half << 3), 32);
            acc0[j] *= rc; acc1[j] *= rc; acc2[j] *= rc; acc3[j] *= rc;
        }

        // P is already in A-fragment layout (the S^T trick): just pack to bf16.
        v16bf pa;
#pragma unroll
        for (int i = 0; i < 8; ++i) {
            pa[i]     = (__bf16)p0[i];
            pa[i + 8] = (__bf16)p1[i];
        }

        // ---- out += P x V : 4 dv column tiles, contraction K = 32 keys ----
#pragma unroll
        for (int t4 = 0; t4 < 4; ++t4) {
            // B-fragment: lane col dv = t4*16+l16, K = keys {half*8..+7, 16+half*8..+7}
            // -> two 8-wide bf16 runs from the transposed V tile (matches P).
            const __bf16* vr = vb_base + (t4 * 16 + l16) * kVP + half * 8;
            v16bf vbf;
#pragma unroll
            for (int i = 0; i < 16; ++i) vbf[i] = vr[i < 8 ? i : i + 8];
            v8f& a = (t4 == 0) ? acc0 : (t4 == 1) ? acc1 : (t4 == 2) ? acc2 : acc3;
            a = __builtin_amdgcn_wmma_f32_16x16x32_bf16(false, pa, false, vbf,
                                                        (short)0, a, false, false);
        }
    }

    // ---- epilogue: normalize by row sum and store ----
    const float linv = 1.0f / lrun;
#pragma unroll
    for (int j = 0; j < 8; ++j) {
        const float rl = __shfl(linv, j + (half << 3), 32);
        float* orow = Oh + (size_t)(q0 + j + half * 8) * kDV;
        orow[ 0 + l16] = acc0[j] * rl;
        orow[16 + l16] = acc1[j] * rl;
        orow[32 + l16] = acc2[j] * rl;
        orow[48 + l16] = acc3[j] * rl;
    }
}

extern "C" void kernel_launch(void* const* d_in, const int* in_sizes, int n_in,
                              void* d_out, int out_size, void* d_ws, size_t ws_size,
                              hipStream_t stream) {
    (void)n_in; (void)out_size; (void)d_ws; (void)ws_size;
    // setup_inputs order: key, query, value  (all fp32)
    const float* Kd = (const float*)d_in[0];
    const float* Qd = (const float*)d_in[1];
    const float* Vd = (const float*)d_in[2];
    float* Od = (float*)d_out;

    const int B = in_sizes[0] / (kH * kS * kDK);   // = 2
    dim3 grid(kS / 64, kH, B);
    dim3 block(128);
    fa_fwd_kernel<<<grid, block, 0, stream>>>(Kd, Qd, Vd, Od);
}
